// Attention_16492674417261
// MI455X (gfx1250) — matrix operand order repaired and checked by reading, not compiled, not run on hardware
//
#include <hip/hip_runtime.h>
#include <hip/hip_bf16.h>

// ---- problem constants (reference) ----
#define D_MODEL   1024
#define NUM_HEADS 16
#define DK        64
#define SEQ       2048
#define BATCH     4

typedef __bf16 bf16;
typedef __attribute__((ext_vector_type(16))) __bf16 v16bf;
typedef __attribute__((ext_vector_type(8)))  float  v8f;

union Frag { v16bf v; int4 q[2]; };

__device__ __forceinline__ bf16 f2bf(float x) {
  union { float f; unsigned u; } c; c.f = x;
  unsigned r = (c.u + 0x7FFFu + ((c.u >> 16) & 1u)) >> 16;  // RNE
  union { unsigned short s; bf16 b; } o; o.s = (unsigned short)r;
  return o.b;
}

// A-fragment (16x32 bf16, row-major, lane = row, kb = 0/8 by lane-half)
__device__ __forceinline__ v16bf load_frag_a(const bf16* src, int ld) {
  const int lane = threadIdx.x & 31;
  const int r    = lane & 15;
  const int kb   = (lane < 16) ? 0 : 8;
  Frag f;
  f.q[0] = *(const int4*)(src + (size_t)r * ld + kb);        // k = kb..kb+7
  f.q[1] = *(const int4*)(src + (size_t)r * ld + kb + 16);   // k = kb+16..kb+23
  return f.v;
}

// B-fragment (32x16 bf16) from a TRANSPOSED source Bt[n][k]
// lane = output column n, contraction k contiguous; kb = 0/8 by lane-half
__device__ __forceinline__ v16bf load_frag_bt(const bf16* src, int ld) {
  const int lane = threadIdx.x & 31;
  const int n    = lane & 15;
  const int kb   = (lane < 16) ? 0 : 8;
  Frag f;
  f.q[0] = *(const int4*)(src + (size_t)n * ld + kb);        // k = kb..kb+7
  f.q[1] = *(const int4*)(src + (size_t)n * ld + kb + 16);   // k = kb+16..kb+23
  return f.v;
}

__device__ __forceinline__ v8f wmma_bf16(v16bf a, v16bf b, v8f c) {
  return __builtin_amdgcn_wmma_f32_16x16x32_bf16(
      /*neg_a=*/false, a, /*neg_b=*/false, b,
      /*c_mod=*/(short)0, c, /*reuse_a=*/false, /*reuse_b=*/false);
}

// ---------------- LayerNorm -> bf16 ----------------
__global__ __launch_bounds__(256) void ln_kernel(
    const float* __restrict__ x, const float* __restrict__ gamma,
    const float* __restrict__ beta, bf16* __restrict__ xn) {
  __shared__ float sred[16];
  const int row = blockIdx.x;
  const int t = threadIdx.x;
  const float4 v = ((const float4*)(x + (size_t)row * D_MODEL))[t];
  float s  = v.x + v.y + v.z + v.w;
  float ss = v.x * v.x + v.y * v.y + v.z * v.z + v.w * v.w;
#pragma unroll
  for (int off = 16; off >= 1; off >>= 1) {
    s  += __shfl_xor(s,  off, 32);
    ss += __shfl_xor(ss, off, 32);
  }
  const int wave = t >> 5, lane = t & 31;
  if (lane == 0) { sred[wave] = s; sred[8 + wave] = ss; }
  __syncthreads();
  float ts = 0.f, tss = 0.f;
#pragma unroll
  for (int w = 0; w < 8; ++w) { ts += sred[w]; tss += sred[8 + w]; }
  const float mu  = ts * (1.0f / D_MODEL);
  const float var = tss * (1.0f / D_MODEL) - mu * mu;
  const float r   = rsqrtf(var + 1e-5f);
  const float4 g  = ((const float4*)gamma)[t];
  const float4 be = ((const float4*)beta)[t];
  bf16* orow = xn + (size_t)row * D_MODEL + t * 4;
  orow[0] = f2bf((v.x - mu) * r * g.x + be.x);
  orow[1] = f2bf((v.y - mu) * r * g.y + be.y);
  orow[2] = f2bf((v.z - mu) * r * g.z + be.z);
  orow[3] = f2bf((v.w - mu) * r * g.w + be.w);
}

// ------- weight f32[k][n] -> bf16 transposed Wt[n][k] -------
__global__ __launch_bounds__(256) void convt_kernel(
    const float* __restrict__ W, bf16* __restrict__ Wt) {
  const int idx = blockIdx.x * 256 + threadIdx.x;   // idx = k*1024 + n
  const int k = idx >> 10, n = idx & 1023;
  Wt[(size_t)n * 1024 + k] = f2bf(W[idx]);
}

// ---------------- generic bf16 WMMA GEMM ----------------
// C[m][n] = A[m][:] . Bt[n][:] + bias[n]; epilogue by mode:
//   0: bf16 out laid out [b,h,s,d]      (Q and K)
//   1: bf16 out laid out [b,h,d,s]      (V transposed for the PV stage)
//   2: f32 out = acc + bias + residual  (output projection)
__global__ __launch_bounds__(256) void gemm_kernel(
    const bf16* __restrict__ A, const bf16* __restrict__ Bt,
    const float* __restrict__ bias, const float* __restrict__ residual,
    void* __restrict__ out, int N, int Kdim, int mode) {
  const int wave = threadIdx.x >> 5;
  const int lane = threadIdx.x & 31;
  const int wm = wave & 3, wn = wave >> 2;
  const int m0 = blockIdx.x * 128 + wm * 32;
  const int n0 = blockIdx.y * 128 + wn * 64;
  v8f acc[2][4] = {};
  const bf16* Arow0 = A + (size_t)m0 * Kdim;
  const bf16* Arow1 = A + (size_t)(m0 + 16) * Kdim;
  for (int k = 0; k < Kdim; k += 32) {
    __builtin_prefetch(Arow0 + k + 256, 0, 1);   // global_prefetch_b8
    v16bf a0 = load_frag_a(Arow0 + k, Kdim);
    v16bf a1 = load_frag_a(Arow1 + k, Kdim);
#pragma unroll
    for (int ni = 0; ni < 4; ++ni) {
      v16bf b = load_frag_bt(Bt + (size_t)(n0 + ni * 16) * Kdim + k, Kdim);
      acc[0][ni] = wmma_bf16(a0, b, acc[0][ni]);
      acc[1][ni] = wmma_bf16(a1, b, acc[1][ni]);
    }
  }
  // epilogue: C-fragment layout row = i + 8*half, col = lane&15
  const int col = lane & 15;
  const int rb  = (lane < 16) ? 0 : 8;
#pragma unroll
  for (int mi = 0; mi < 2; ++mi)
#pragma unroll
    for (int ni = 0; ni < 4; ++ni)
#pragma unroll
      for (int i = 0; i < 8; ++i) {
        const int m = m0 + mi * 16 + rb + i;
        const int n = n0 + ni * 16 + col;
        float v = acc[mi][ni][i] + bias[n];
        if (mode == 2) {
          ((float*)out)[(size_t)m * N + n] = v + residual[(size_t)m * N + n];
        } else {
          const int b = m >> 11, s = m & 2047;   // SEQ = 2048
          const int h = n >> 6,  d = n & 63;     // DK  = 64
          bf16* ob = (bf16*)out;
          if (mode == 0)
            ob[(((size_t)b * NUM_HEADS + h) * SEQ + s) * DK + d] = f2bf(v);
          else
            ob[(((size_t)b * NUM_HEADS + h) * DK + d) * SEQ + s] = f2bf(v);
        }
      }
}

// ---------------- flash attention (causal) ----------------
// grid: (S/128, B*H); 8 waves; wave w owns q rows [q0, q0+16)
__global__ __launch_bounds__(256) void attn_kernel(
    const bf16* __restrict__ Q, const bf16* __restrict__ Kmat,
    const bf16* __restrict__ Vt, bf16* __restrict__ O) {
  __shared__ bf16 pbuf[8][16 * 40];          // per-wave P tile, pitch 40 (16B-aligned rows)
  const int wave = threadIdx.x >> 5;
  const int lane = threadIdx.x & 31;
  const int col  = lane & 15;
  const int rb   = (lane < 16) ? 0 : 8;
  const int bh   = blockIdx.y;
  const int q0   = blockIdx.x * 128 + wave * 16;
  const bf16* Qb = Q    + (size_t)bh * SEQ * DK;
  const bf16* Kb = Kmat + (size_t)bh * SEQ * DK;
  const bf16* Vb = Vt   + (size_t)bh * DK * SEQ;

  const v16bf aq0 = load_frag_a(Qb + (size_t)q0 * DK,      DK);  // d = 0..31
  const v16bf aq1 = load_frag_a(Qb + (size_t)q0 * DK + 32, DK);  // d = 32..63

  v8f acc[4] = {};
  float m_i[8], l_i[8];
#pragma unroll
  for (int i = 0; i < 8; ++i) { m_i[i] = -__builtin_inff(); l_i[i] = 0.f; }

  bf16* pw = &pbuf[wave][0];
  const int kv_end = blockIdx.x * 128 + 128;   // uniform per block (masking covers rest)

  for (int kv = 0; kv < kv_end; kv += 32) {
    // scores: two 16-key WMMA tiles, contraction over d=64 in two steps
    v8f s0 = {}, s1 = {};
    {
      v16bf b00 = load_frag_bt(Kb + (size_t)kv * DK,      DK);
      v16bf b01 = load_frag_bt(Kb + (size_t)kv * DK + 32, DK);
      s0 = wmma_bf16(aq0, b00, s0);
      s0 = wmma_bf16(aq1, b01, s0);
      v16bf b10 = load_frag_bt(Kb + (size_t)(kv + 16) * DK,      DK);
      v16bf b11 = load_frag_bt(Kb + (size_t)(kv + 16) * DK + 32, DK);
      s1 = wmma_bf16(aq0, b10, s1);
      s1 = wmma_bf16(aq1, b11, s1);
    }
    // online softmax in C-fragment layout; xor masks 1..8 stay within lane-half
#pragma unroll
    for (int i = 0; i < 8; ++i) {
      const int qrow = q0 + rb + i;
      float v0 = s0[i] * 0.125f;                        // 1/sqrt(64)
      float v1 = s1[i] * 0.125f;
      if (kv + col      > qrow) v0 = -__builtin_inff(); // causal mask
      if (kv + 16 + col > qrow) v1 = -__builtin_inff();
      float mx = fmaxf(v0, v1);
#pragma unroll
      for (int off = 8; off >= 1; off >>= 1)
        mx = fmaxf(mx, __shfl_xor(mx, off, 32));
      const float mn    = fmaxf(m_i[i], mx);
      const float alpha = __expf(m_i[i] - mn);
      const float p0 = __expf(v0 - mn);
      const float p1 = __expf(v1 - mn);
      float ps = p0 + p1;
#pragma unroll
      for (int off = 8; off >= 1; off >>= 1)
        ps += __shfl_xor(ps, off, 32);
      l_i[i] = l_i[i] * alpha + ps;
      m_i[i] = mn;
#pragma unroll
      for (int t = 0; t < 4; ++t) acc[t][i] = acc[t][i] * alpha;
      pw[(rb + i) * 40 + col]      = f2bf(p0);
      pw[(rb + i) * 40 + 16 + col] = f2bf(p1);
    }
    asm volatile("s_wait_dscnt 0" ::: "memory");
    const v16bf ap = load_frag_a(pw, 40);                // P as A-fragment (16x32)
#pragma unroll
    for (int t = 0; t < 4; ++t) {
      v16bf bv = load_frag_bt(Vb + (size_t)(t * 16) * SEQ + kv, SEQ);
      acc[t] = wmma_bf16(ap, bv, acc[t]);
    }
    asm volatile("s_wait_dscnt 0" ::: "memory");
  }

  // write attn-out as [B, S, H*DK] bf16 for the output projection
  const int b = bh >> 4, h = bh & 15;
#pragma unroll
  for (int i = 0; i < 8; ++i) {
    const float inv = 1.0f / l_i[i];
    const size_t rowoff = ((size_t)b * SEQ + (q0 + rb + i)) * (NUM_HEADS * DK);
#pragma unroll
    for (int t = 0; t < 4; ++t)
      O[rowoff + h * DK + t * 16 + col] = f2bf(acc[t][i] * inv);
  }
}

// ---------------- host orchestration ----------------
extern "C" void kernel_launch(void* const* d_in, const int* in_sizes, int n_in,
                              void* d_out, int out_size, void* d_ws, size_t ws_size,
                              hipStream_t stream) {
  (void)in_sizes; (void)n_in; (void)out_size; (void)ws_size;
  const float* x     = (const float*)d_in[0];
  const float* Wq    = (const float*)d_in[1];
  const float* bq    = (const float*)d_in[2];
  const float* Wk    = (const float*)d_in[3];
  const float* bk    = (const float*)d_in[4];
  const float* Wv    = (const float*)d_in[5];
  const float* bv    = (const float*)d_in[6];
  const float* Wo    = (const float*)d_in[7];
  const float* bo    = (const float*)d_in[8];
  const float* gamma = (const float*)d_in[9];
  const float* beta  = (const float*)d_in[10];

  const size_t M = (size_t)BATCH * SEQ;         // 8192
  bf16* xn  = (bf16*)d_ws;                      // [8192,1024]
  bf16* WqT = xn  + M * D_MODEL;                // [1024,1024] transposed
  bf16* WkT = WqT + (size_t)D_MODEL * D_MODEL;
  bf16* WvT = WkT + (size_t)D_MODEL * D_MODEL;
  bf16* WoT = WvT + (size_t)D_MODEL * D_MODEL;
  bf16* Qb  = WoT + (size_t)D_MODEL * D_MODEL;  // [B,H,S,DK]
  bf16* Kb  = Qb  + M * D_MODEL;                // [B,H,S,DK]
  bf16* Vt  = Kb  + M * D_MODEL;                // [B,H,DK,S]
  bf16* Oat = Vt  + M * D_MODEL;                // [B,S,H*DK]

  ln_kernel<<<(int)M, 256, 0, stream>>>(x, gamma, beta, xn);
  convt_kernel<<<4096, 256, 0, stream>>>(Wq, WqT);
  convt_kernel<<<4096, 256, 0, stream>>>(Wk, WkT);
  convt_kernel<<<4096, 256, 0, stream>>>(Wv, WvT);
  convt_kernel<<<4096, 256, 0, stream>>>(Wo, WoT);

  dim3 g(64, 8), blk(256);
  gemm_kernel<<<g, blk, 0, stream>>>(xn, WqT, bq, nullptr, Qb, D_MODEL, D_MODEL, 0);
  gemm_kernel<<<g, blk, 0, stream>>>(xn, WkT, bk, nullptr, Kb, D_MODEL, D_MODEL, 0);
  gemm_kernel<<<g, blk, 0, stream>>>(xn, WvT, bv, nullptr, Vt, D_MODEL, D_MODEL, 1);

  attn_kernel<<<dim3(SEQ / 128, BATCH * NUM_HEADS), blk, 0, stream>>>(Qb, Kb, Vt, Oat);

  gemm_kernel<<<g, blk, 0, stream>>>(Oat, WoT, bo, x, d_out, D_MODEL, D_MODEL, 2);
}
